// GFootballMamba_6313601925547
// MI455X (gfx1250) — matrix operand order repaired and checked by reading, not compiled, not run on hardware
//
#include <hip/hip_runtime.h>
#include <hip/hip_bf16.h>

typedef __attribute__((ext_vector_type(8)))  float  v8f;
typedef __attribute__((ext_vector_type(16))) __bf16 v16bf;

#define D_MODEL 1024
#define NSTATE  16
#define BATCH   8
#define SEQ     1024
#define ROWS    (BATCH * SEQ)   // 8192
#define LDS_PITCH 40            // ushorts per row: 80B = 20 dwords, 16B aligned,
                                // 20*l mod 64 distinct for l=0..15 -> conflict-free

// ---------------------------------------------------------------- helpers
__device__ __forceinline__ unsigned int pack_bf16(float a, float b) {
    unsigned int ua = __float_as_uint(a);
    unsigned int ub = __float_as_uint(b);
    ua = (ua + 0x7FFFu + ((ua >> 16) & 1u)) >> 16;   // RNE
    ub = (ub + 0x7FFFu + ((ub >> 16) & 1u)) >> 16;
    return (ua & 0xFFFFu) | (ub << 16);
}
__device__ __forceinline__ unsigned short f2bf1(float f) {
    unsigned int u = __float_as_uint(f);
    return (unsigned short)((u + 0x7FFFu + ((u >> 16) & 1u)) >> 16);
}

union Frag { unsigned int u[8]; v16bf v; };

// ---------------------------------------------------------------- layernorm -> bf16
__global__ __launch_bounds__(256)
void ln_kernel(const float* __restrict__ x, const float* __restrict__ g,
               const float* __restrict__ b, unsigned short* __restrict__ xn_bf) {
    __shared__ float rs[8], rss[8];
    int row = blockIdx.x;
    const float4* xr = (const float4*)(x + (size_t)row * D_MODEL);
    float4 v = xr[threadIdx.x];
    float s  = v.x + v.y + v.z + v.w;
    float ss = v.x*v.x + v.y*v.y + v.z*v.z + v.w*v.w;
    #pragma unroll
    for (int m = 16; m; m >>= 1) { s += __shfl_xor(s, m, 32); ss += __shfl_xor(ss, m, 32); }
    int wid = threadIdx.x >> 5;
    if ((threadIdx.x & 31) == 0) { rs[wid] = s; rss[wid] = ss; }
    __syncthreads();
    float tot = 0.f, tot2 = 0.f;
    #pragma unroll
    for (int i = 0; i < 8; i++) { tot += rs[i]; tot2 += rss[i]; }
    float mu  = tot  * (1.0f / D_MODEL);
    float var = tot2 * (1.0f / D_MODEL) - mu * mu;
    float inv = rsqrtf(var + 1e-5f);
    float4 gv = ((const float4*)g)[threadIdx.x];
    float4 bv = ((const float4*)b)[threadIdx.x];
    float ox = (v.x - mu) * inv * gv.x + bv.x;
    float oy = (v.y - mu) * inv * gv.y + bv.y;
    float oz = (v.z - mu) * inv * gv.z + bv.z;
    float ow = (v.w - mu) * inv * gv.w + bv.w;
    uint2 packed;
    packed.x = pack_bf16(ox, oy);
    packed.y = pack_bf16(oz, ow);
    ((uint2*)(xn_bf + (size_t)row * D_MODEL))[threadIdx.x] = packed;
}

// ---------------------------------------------------------------- f32 -> bf16 weight convert
__global__ __launch_bounds__(256)
void cvt_kernel(const float* __restrict__ s, unsigned short* __restrict__ d, int n4) {
    int i = blockIdx.x * 256 + threadIdx.x;
    if (i < n4) {
        float4 v = ((const float4*)s)[i];
        uint2 p;
        p.x = pack_bf16(v.x, v.y);
        p.y = pack_bf16(v.z, v.w);
        ((uint2*)d)[i] = p;
    }
}

// ---------------------------------------------------------------- WMMA GEMM (bf16 operands)
// C[M,N] = A[M,K] * W[N,K]^T.  MODE 0: split f32 write x_in/z + bf16 copy of
// x_in (uniform per block).  MODE 1: +bias, softplus.  MODE 2: +residual.
// Double-buffered LDS; staging is a pure b128 copy (no VALU conversion).
template <int MODE>
__global__ __launch_bounds__(256)
void gemm_wmma(const unsigned short* __restrict__ A, const unsigned short* __restrict__ W,
               float* __restrict__ out0, float* __restrict__ out1,
               unsigned short* __restrict__ out_bf,
               const float* __restrict__ bias, const float* __restrict__ resid,
               int M, int N, int K) {
    __shared__ unsigned short As[2][128 * LDS_PITCH];   // [m][k]
    __shared__ unsigned short Bs[2][128 * LDS_PITCH];   // [n][k]

    const int tid  = threadIdx.x;
    const int lane = tid & 31;
    const int wid  = tid >> 5;
    const int wm   = wid & 1;          // 2 waves along M (64 rows each)
    const int wn   = wid >> 1;         // 4 waves along N (32 cols each)
    const int block_m = blockIdx.y * 128;
    const int block_n = blockIdx.x * 128;

    const int ld_r = tid >> 1;         // 0..127
    const int ld_k = (tid & 1) * 16;   // 0 or 16 (k elements)

    const int l15 = lane & 15;
    const int hi  = lane >> 4;

    const unsigned short* aldp = A + (size_t)(block_m + ld_r) * K + ld_k;
    const unsigned short* wldp = W + (size_t)(block_n + ld_r) * K + ld_k;

    v8f acc[4][2];
    #pragma unroll
    for (int i = 0; i < 4; i++)
        #pragma unroll
        for (int j = 0; j < 2; j++)
            acc[i][j] = (v8f){0.f,0.f,0.f,0.f,0.f,0.f,0.f,0.f};

    const int NT = K >> 5;             // K-steps of 32

    // ---- prologue: stage tile 0 into buffer 0 (pure copy)
    {
        const uint4* ap = (const uint4*)aldp;
        const uint4* wp = (const uint4*)wldp;
        uint4 a0 = ap[0], a1 = ap[1];
        uint4 b0 = wp[0], b1 = wp[1];
        uint4* as = (uint4*)&As[0][ld_r * LDS_PITCH + ld_k];
        uint4* bs = (uint4*)&Bs[0][ld_r * LDS_PITCH + ld_k];
        as[0] = a0; as[1] = a1;
        bs[0] = b0; bs[1] = b1;
    }

    for (int ks = 0; ks < NT; ks++) {
        __syncthreads();               // buffer `cur` ready; buffer `nxt` free
        const int cur = ks & 1;
        const int nxt = cur ^ 1;
        const bool more = (ks + 1) < NT;

        // ---- issue next tile's global loads first (overlap with WMMA)
        uint4 a0, a1, b0, b1;
        if (more) {
            const int k1 = (ks + 1) << 5;
            const uint4* ap = (const uint4*)(aldp + k1);
            const uint4* wp = (const uint4*)(wldp + k1);
            a0 = ap[0]; a1 = ap[1];
            b0 = wp[0]; b1 = wp[1];
        }

        // ---- B fragments for this wave's 2 N-tiles
        Frag bfr[2];
        #pragma unroll
        for (int tn = 0; tn < 2; tn++) {
            const int nrow = wn * 32 + tn * 16 + l15;
            #pragma unroll
            for (int vv = 0; vv < 8; vv++) {
                const int kk = 2 * vv + ((vv >= 4) ? 8 : 0) + hi * 8;  // B layout (same as A)
                bfr[tn].u[vv] = *(const unsigned int*)&Bs[cur][nrow * LDS_PITCH + kk];
            }
        }
        // ---- A fragments + WMMA
        #pragma unroll
        for (int tm = 0; tm < 4; tm++) {
            const int mrow = wm * 64 + tm * 16 + l15;
            Frag af;
            #pragma unroll
            for (int vv = 0; vv < 8; vv++) {
                const int kk = 2 * vv + ((vv >= 4) ? 8 : 0) + hi * 8;  // A layout
                af.u[vv] = *(const unsigned int*)&As[cur][mrow * LDS_PITCH + kk];
            }
            #pragma unroll
            for (int tn = 0; tn < 2; tn++) {
                acc[tm][tn] = __builtin_amdgcn_wmma_f32_16x16x32_bf16(
                    false, af.v, false, bfr[tn].v, (short)0, acc[tm][tn], false, false);
            }
        }

        // ---- store next tile into the other buffer (visible after next barrier)
        if (more) {
            uint4* as = (uint4*)&As[nxt][ld_r * LDS_PITCH + ld_k];
            uint4* bs = (uint4*)&Bs[nxt][ld_r * LDS_PITCH + ld_k];
            as[0] = a0; as[1] = a1;
            bs[0] = b0; bs[1] = b1;
        }
    }

    // ---- epilogue: C/D layout M = vv + 8*hi, N = lane&15
    const bool lo_half = (block_n < D_MODEL);   // uniform per block for MODE 0
    #pragma unroll
    for (int tm = 0; tm < 4; tm++) {
        #pragma unroll
        for (int tn = 0; tn < 2; tn++) {
            const int col = block_n + wn * 32 + tn * 16 + l15;
            float badd = (MODE == 1) ? bias[col] : 0.f;
            #pragma unroll
            for (int vv = 0; vv < 8; vv++) {
                const int row = block_m + wm * 64 + tm * 16 + vv + 8 * hi;
                float val = acc[tm][tn][vv];
                if (MODE == 0) {
                    if (lo_half) {
                        out0[(size_t)row * D_MODEL + col] = val;                 // x_in f32
                        out_bf[(size_t)row * D_MODEL + col] = f2bf1(val);        // x_in bf16
                    } else {
                        out1[(size_t)row * D_MODEL + (col - D_MODEL)] = val;     // z f32
                    }
                } else if (MODE == 1) {
                    val += badd;
                    val = (val > 20.f) ? val : log1pf(__expf(val));   // softplus
                    out0[(size_t)row * D_MODEL + col] = val;
                } else {
                    val += resid[(size_t)row * D_MODEL + col];
                    out0[(size_t)row * D_MODEL + col] = val;
                }
            }
        }
    }
}

// ---------------------------------------------------------------- B_t / C_t projections (N=16)
__global__ __launch_bounds__(256)
void bc_kernel(const float* __restrict__ x_in, const float* __restrict__ W_B,
               const float* __restrict__ W_C, float* __restrict__ Bt,
               float* __restrict__ Ct) {
    int gtid = blockIdx.x * 256 + threadIdx.x;   // ROWS*32 threads
    int row  = gtid >> 5;
    int sub  = gtid & 31;
    int n    = sub & 15;
    const float4* w  = (const float4*)(((sub < 16) ? W_B : W_C) + (size_t)n * D_MODEL);
    const float4* xr = (const float4*)(x_in + (size_t)row * D_MODEL);
    float acc = 0.f;
    #pragma unroll 4
    for (int k = 0; k < D_MODEL / 4; k++) {
        float4 xv = xr[k];
        float4 wv = w[k];
        acc += xv.x * wv.x + xv.y * wv.y + xv.z * wv.z + xv.w * wv.w;
    }
    float* dst = (sub < 16) ? Bt : Ct;
    dst[(size_t)row * NSTATE + n] = acc;
}

// ---------------------------------------------------------------- sequential SSM scan
// thread = (b, d, n); 16-lane butterfly reduce over n; fused silu/D epilogue,
// ssm written as bf16 (only consumer is GEMM-2's A operand).
__global__ __launch_bounds__(256)
void scan_kernel(const float* __restrict__ dt, const float* __restrict__ x_in,
                 const float* __restrict__ Bt, const float* __restrict__ Ct,
                 const float* __restrict__ z, const float* __restrict__ A_log,
                 const float* __restrict__ h_prev, const float* __restrict__ Dv,
                 unsigned short* __restrict__ ssm_bf, float* __restrict__ h_final) {
    int gtid = blockIdx.x * 256 + threadIdx.x;   // BATCH*D_MODEL*NSTATE threads
    int n   = gtid & 15;
    int ch  = gtid >> 4;            // 0..8191
    int b   = ch >> 10;
    int dch = ch & 1023;

    float a  = -__expf(A_log[dch * NSTATE + n]);
    float h  = h_prev[(size_t)b * D_MODEL * NSTATE + dch * NSTATE + n];
    float dd = Dv[dch];

    const float* dtp = dt   + (size_t)b * SEQ * D_MODEL + dch;
    const float* xp  = x_in + (size_t)b * SEQ * D_MODEL + dch;
    const float* zp  = z    + (size_t)b * SEQ * D_MODEL + dch;
    const float* bp  = Bt   + (size_t)b * SEQ * NSTATE  + n;
    const float* cp  = Ct   + (size_t)b * SEQ * NSTATE  + n;
    unsigned short* sp = ssm_bf + (size_t)b * SEQ * D_MODEL + dch;

    for (int t = 0; t < SEQ; t++) {
        if (t + 16 < SEQ) {                       // global_prefetch_b8 ahead
            __builtin_prefetch(&dtp[(size_t)(t + 16) * D_MODEL], 0, 0);
            __builtin_prefetch(&xp [(size_t)(t + 16) * D_MODEL], 0, 0);
            __builtin_prefetch(&zp [(size_t)(t + 16) * D_MODEL], 0, 0);
        }
        float dtv = dtp[(size_t)t * D_MODEL];
        float xv  = xp [(size_t)t * D_MODEL];
        float btv = bp [(size_t)t * NSTATE];
        float ctv = cp [(size_t)t * NSTATE];
        h = __expf(dtv * a) * h + dtv * xv * btv;
        float p = h * ctv;
        p += __shfl_xor(p, 1, 16);
        p += __shfl_xor(p, 2, 16);
        p += __shfl_xor(p, 4, 16);
        p += __shfl_xor(p, 8, 16);
        if (n == 0) {
            float zv = zp[(size_t)t * D_MODEL];
            float sil = zv / (1.f + __expf(-zv));
            sp[(size_t)t * D_MODEL] = f2bf1(p * sil + xv * dd);
        }
    }
    h_final[(size_t)b * D_MODEL * NSTATE + dch * NSTATE + n] = h;
}

// ---------------------------------------------------------------- launch
extern "C" void kernel_launch(void* const* d_in, const int* in_sizes, int n_in,
                              void* d_out, int out_size, void* d_ws, size_t ws_size,
                              hipStream_t stream) {
    const float* x     = (const float*)d_in[0];
    const float* hprev = (const float*)d_in[1];
    const float* ln_g  = (const float*)d_in[2];
    const float* ln_b  = (const float*)d_in[3];
    const float* W_ig  = (const float*)d_in[4];
    const float* W_dt  = (const float*)d_in[5];
    const float* b_dt  = (const float*)d_in[6];
    const float* A_log = (const float*)d_in[7];
    const float* W_B   = (const float*)d_in[8];
    const float* W_C   = (const float*)d_in[9];
    const float* Dv    = (const float*)d_in[10];
    const float* W_out = (const float*)d_in[11];
    float* out = (float*)d_out;

    // ---- workspace carve-up (bytes)
    char* p = (char*)d_ws;
    const size_t BIGF = (size_t)ROWS * D_MODEL * sizeof(float);     // 32 MB
    const size_t BIGH = (size_t)ROWS * D_MODEL * sizeof(short);     // 16 MB
    unsigned short* xn_bf   = (unsigned short*)p;  p += BIGH;
    unsigned short* Wig_bf  = (unsigned short*)p;  p += (size_t)2 * D_MODEL * D_MODEL * 2;
    unsigned short* Wdt_bf  = (unsigned short*)p;  p += (size_t)D_MODEL * D_MODEL * 2;
    unsigned short* Wout_bf = (unsigned short*)p;  p += (size_t)D_MODEL * D_MODEL * 2;
    float*          x_in    = (float*)p;           p += BIGF;
    unsigned short* x_in_bf = (unsigned short*)p;  p += BIGH;
    float*          zbuf    = (float*)p;           p += BIGF;
    float*          dt      = (float*)p;           p += BIGF;
    unsigned short* ssm_bf  = (unsigned short*)p;  p += BIGH;
    float*          Bt      = (float*)p;           p += (size_t)ROWS * NSTATE * sizeof(float);
    float*          Ct      = (float*)p;

    // ---- one-time (per launch) bf16 conversions
    ln_kernel<<<ROWS, 256, 0, stream>>>(x, ln_g, ln_b, xn_bf);
    cvt_kernel<<<(2 * D_MODEL * D_MODEL / 4 + 255) / 256, 256, 0, stream>>>(
        W_ig, Wig_bf, 2 * D_MODEL * D_MODEL / 4);
    cvt_kernel<<<(D_MODEL * D_MODEL / 4 + 255) / 256, 256, 0, stream>>>(
        W_dt, Wdt_bf, D_MODEL * D_MODEL / 4);
    cvt_kernel<<<(D_MODEL * D_MODEL / 4 + 255) / 256, 256, 0, stream>>>(
        W_out, Wout_bf, D_MODEL * D_MODEL / 4);

    gemm_wmma<0><<<dim3(2 * D_MODEL / 128, ROWS / 128), 256, 0, stream>>>(
        xn_bf, Wig_bf, x_in, zbuf, x_in_bf, nullptr, nullptr, ROWS, 2 * D_MODEL, D_MODEL);

    gemm_wmma<1><<<dim3(D_MODEL / 128, ROWS / 128), 256, 0, stream>>>(
        x_in_bf, Wdt_bf, dt, nullptr, nullptr, b_dt, nullptr, ROWS, D_MODEL, D_MODEL);

    bc_kernel<<<(ROWS * 32) / 256, 256, 0, stream>>>(x_in, W_B, W_C, Bt, Ct);

    scan_kernel<<<(ROWS * NSTATE) / 256, 256, 0, stream>>>(
        dt, x_in, Bt, Ct, zbuf, A_log, hprev, Dv, ssm_bf,
        out + (size_t)ROWS * D_MODEL);

    gemm_wmma<2><<<dim3(D_MODEL / 128, ROWS / 128), 256, 0, stream>>>(
        ssm_bf, Wout_bf, out, nullptr, nullptr, nullptr, x, ROWS, D_MODEL, D_MODEL);
}
